// Feature_Attention_77300821394210
// MI455X (gfx1250) — matrix operand order repaired and checked by reading, not compiled, not run on hardware
//
#include <hip/hip_runtime.h>
#include <hip/hip_bf16.h>
#include <math.h>

// Problem dims
#define S_   4
#define B_   16
#define C_   256
#define H_   64
#define W_   64
#define HW_  4096
#define SB_  64      // S_*B_

// Workspace layout (floats)
#define WS_FBAR  0                    // SB_*C_ = 16384
#define WS_G     16384                // SB_*C_ = 16384
#define WS_S0    32768                // SB_     = 64
#define WS_GATE  32832                // SB_*HW_ = 262144

// LDS strides (bf16 elements) chosen for bank spread + 16B alignment
#define GSTR 72                       // gate row stride   (144B)
#define WSTR 264                      // Wv row stride     (528B)
#define FSTR 1032                     // fB row stride     (2064B)
#define VK   24                       // vB k-row stride   (48B)
#define VC   (64 * VK)                // vB channel stride (1536 bf16)

typedef __attribute__((ext_vector_type(16))) __bf16 v16bf;
typedef __attribute__((ext_vector_type(8)))  float  v8f;

// ---------------------------------------------------------------------------
// Kernel 1: fbar[sb,c] = mean over hw of features[sb,c,:]   (float4 pass)
// ---------------------------------------------------------------------------
__global__ void k_fbar(const float* __restrict__ f, float* __restrict__ ws) {
    __shared__ float red[256];
    const int blk = blockIdx.x;           // sb*C_ + c  (matches f layout)
    const int tid = threadIdx.x;
    const float4* src = (const float4*)(f + (size_t)blk * HW_);
    float acc = 0.f;
#pragma unroll
    for (int i = 0; i < HW_ / 1024; ++i) {
        float4 v = src[tid + i * 256];
        acc += (v.x + v.y) + (v.z + v.w);
    }
    red[tid] = acc;
    __syncthreads();
    for (int s = 128; s > 0; s >>= 1) {
        if (tid < s) red[tid] += red[tid + s];
        __syncthreads();
    }
    if (tid == 0) ws[WS_FBAR + blk] = red[0] * (1.f / (float)HW_);
}

// ---------------------------------------------------------------------------
// Kernel 2: per (s,b): kbar = Wk@fbar + bk;  g = Wq^T @ kbar;  s0 = bq.kbar
// ---------------------------------------------------------------------------
__global__ void k_prep(const float* __restrict__ Wq, const float* __restrict__ bq,
                       const float* __restrict__ Wk, const float* __restrict__ bk,
                       float* __restrict__ ws) {
    __shared__ float fbar_l[C_];
    __shared__ float kbar_l[C_];
    __shared__ float red[C_];
    const int sb = blockIdx.x;
    const int s  = sb >> 4;
    const int c  = threadIdx.x;

    fbar_l[c] = ws[WS_FBAR + sb * C_ + c];
    __syncthreads();

    float kacc = bk[s * C_ + c];
    const float* wkrow = Wk + ((size_t)(s * C_ + c)) * C_;
#pragma unroll 4
    for (int i = 0; i < C_; ++i) kacc += wkrow[i] * fbar_l[i];
    kbar_l[c] = kacc;
    red[c] = bq[s * C_ + c] * kacc;
    __syncthreads();

    float gacc = 0.f;
#pragma unroll 4
    for (int cc = 0; cc < C_; ++cc)
        gacc += Wq[((size_t)(s * C_ + cc)) * C_ + c] * kbar_l[cc];
    ws[WS_G + sb * C_ + c] = gacc;

    for (int st = 128; st > 0; st >>= 1) {
        if (c < st) red[c] += red[c + st];
        __syncthreads();
    }
    if (c == 0) ws[WS_S0 + sb] = red[0];
}

// ---------------------------------------------------------------------------
// Kernel 3: gate[sb,p] = sigmoid( sum_c f[sb,c,p]*g[sb,c] + s0[sb] )
// float4-vectorized: one block = 1024 px, each thread owns 4 contiguous px.
// ---------------------------------------------------------------------------
__global__ void k_gate(const float* __restrict__ f, float* __restrict__ ws) {
    __shared__ float gl[C_];
    const int blk = blockIdx.x;           // SB_*4 blocks
    const int sb  = blk >> 2;
    const int p4  = (blk & 3) * 256 + threadIdx.x;   // float4 index
    gl[threadIdx.x] = ws[WS_G + sb * C_ + threadIdx.x];
    __syncthreads();
    const float s0 = ws[WS_S0 + sb];
    float ax = s0, ay = s0, az = s0, aw = s0;
    const float4* base = (const float4*)(f + (size_t)sb * C_ * HW_) + p4;
#pragma unroll 4
    for (int c = 0; c < C_; ++c) {
        float4 v = base[(size_t)c * (HW_ / 4)];
        float g = gl[c];
        ax += v.x * g; ay += v.y * g; az += v.z * g; aw += v.w * g;
    }
    float4 o;
    o.x = 1.f / (1.f + __expf(-ax));
    o.y = 1.f / (1.f + __expf(-ay));
    o.z = 1.f / (1.f + __expf(-az));
    o.w = 1.f / (1.f + __expf(-aw));
    ((float4*)(ws + WS_GATE + (size_t)sb * HW_))[p4] = o;
}

// ---------------------------------------------------------------------------
// Kernel 4: fused  v = Wv@f  ->  weights[c] = gate @ v[c]  ->  out = f*weights
// grid = (SB_, W_/16 jtiles, C_/16 cblocks), 512 threads = 16 waves.
// Register-pipelined staging: prefetch f-tile kb+1 (global b128) while the
// matrix pipe consumes tile kb from LDS.
// ---------------------------------------------------------------------------
__global__ void __launch_bounds__(512, 1)
k_main(const float* __restrict__ f, const float* __restrict__ Wv,
       const float* __restrict__ bv, const float* __restrict__ ws,
       float* __restrict__ out) {
    // ~133 KB total LDS
    __shared__ __align__(16) __bf16 gate_bf[64 * GSTR];   //  9.2 KB
    __shared__ float  Rrow[64];
    __shared__ __align__(16) __bf16 Wv_bf[16 * WSTR];     //  8.4 KB
    __shared__ __align__(16) __bf16 fB[32 * FSTR];        // 66.0 KB
    __shared__ __align__(16) __bf16 vB[16 * VC];          // 49.2 KB

    const int sb  = blockIdx.x;
    const int j0  = blockIdx.y * 16;
    const int c0  = blockIdx.z * 16;
    const int s   = sb >> 4;
    const int tid = threadIdx.x;
    const int wave = tid >> 5;         // 0..15
    const int lane = tid & 31;
    const int half = lane >> 4;        // 0/1
    const int l16  = lane & 15;

    const size_t fbase = (size_t)sb * C_ * HW_;
    const float* gate_g = ws + WS_GATE + (size_t)sb * HW_;

    // Per-thread staging geometry: 16 float4 per thread per 32-channel k-block.
    // i4 = tid + t*512 ; m = i4>>8 ; n = (i4&255)*4 ; h = n>>4 ; w = n&15
    int sm[16], sn[16];
    size_t goff[16];
#pragma unroll
    for (int t = 0; t < 16; ++t) {
        int i4 = tid + t * 512;
        int m  = i4 >> 8;
        int n  = (i4 & 255) * 4;
        sm[t] = m; sn[t] = n;
        goff[t] = fbase + (size_t)m * HW_ + (n >> 4) * W_ + j0 + (n & 15);
    }

    // ---- prologue: prefetch k-block 0 + stage gate/Wv/R ----
    float4 pf[16];
#pragma unroll
    for (int t = 0; t < 16; ++t) pf[t] = *(const float4*)(f + goff[t]);

    for (int i = tid; i < 4096; i += 512)
        gate_bf[(i >> 6) * GSTR + (i & 63)] = (__bf16)gate_g[i];
    for (int i = tid; i < 16 * 256; i += 512)
        Wv_bf[(i >> 8) * WSTR + (i & 255)] =
            (__bf16)Wv[((size_t)(s * C_ + c0 + (i >> 8))) * C_ + (i & 255)];
    if (tid < 64) {
        float r = 0.f;
        const float* grow = gate_g + tid * 64;
#pragma unroll
        for (int k = 0; k < 64; ++k) r += grow[k];
        Rrow[tid] = r;
    }
    // store prefetched k-block 0 into fB
#pragma unroll
    for (int t = 0; t < 16; ++t) {
        union { __bf16 h4[4]; uint2 u; } P;
        P.h4[0] = (__bf16)pf[t].x; P.h4[1] = (__bf16)pf[t].y;
        P.h4[2] = (__bf16)pf[t].z; P.h4[3] = (__bf16)pf[t].w;
        *(uint2*)&fB[sm[t] * FSTR + sn[t]] = P.u;
    }
    __syncthreads();

    // ========== Step 1: v_blk[16 x 1024] = Wv_blk @ f_tile ==========
    // M = 16 channels, N = 1024 px (h*16+w), K = 256 (8 blocks of 32)
    v8f acc[4] = {};                       // 4 tiles/wave -> 32 VGPRs
    const int nbase = wave * 64;           // this wave's 4 N-tiles

    for (int kb = 0; kb < 8; ++kb) {
        // prefetch NEXT k-block from global while WMMAs run on this one
        if (kb < 7) {
            const size_t kadv = (size_t)(kb + 1) * 32 * HW_;
#pragma unroll
            for (int t = 0; t < 16; ++t)
                pf[t] = *(const float4*)(f + goff[t] + kadv);
        }

        // A fragment (Wv rows, ISA 16-bit A layout)
        union { v16bf v; unsigned u[8]; } A;
        const unsigned* Wu = (const unsigned*)Wv_bf;
#pragma unroll
        for (int r = 0; r < 8; ++r) {
            int K = (r >> 2) * 16 + half * 8 + (r & 3) * 2;
            A.u[r] = Wu[(l16 * WSTR + kb * 32 + K) >> 1];
        }

        const unsigned* fu = (const unsigned*)fB;
#pragma unroll
        for (int tn = 0; tn < 4; ++tn) {
            union { v16bf v; unsigned u[8]; } Bf;  // B: N col = l16, K per element (table order)
#pragma unroll
            for (int e = 0; e < 16; ++e)
                Bf.v[e] = fB[((e >> 3) * 16 + half * 8 + (e & 7)) * FSTR + nbase + tn * 16 + l16];
            acc[tn] = __builtin_amdgcn_wmma_f32_16x16x32_bf16(
                false, A.v, false, Bf.v, (short)0, acc[tn], false, false);
        }
        __syncthreads();                 // all waves done reading fB

        if (kb < 7) {                    // commit prefetched tile to LDS
#pragma unroll
            for (int t = 0; t < 16; ++t) {
                union { __bf16 h4[4]; uint2 u; } P;
                P.h4[0] = (__bf16)pf[t].x; P.h4[1] = (__bf16)pf[t].y;
                P.h4[2] = (__bf16)pf[t].z; P.h4[3] = (__bf16)pf[t].w;
                *(uint2*)&fB[sm[t] * FSTR + sn[t]] = P.u;
            }
            __syncthreads();
        }
    }

    // write v (bf16) to LDS: C/D layout M = r + 8*half, N = l16
#pragma unroll
    for (int tn = 0; tn < 4; ++tn)
#pragma unroll
        for (int r = 0; r < 8; ++r) {
            int c = r + half * 8;               // 0..15
            int n = nbase + tn * 16 + l16;
            int h = n >> 4, w = n & 15;
            vB[c * VC + h * VK + w] = (__bf16)acc[tn][r];
        }
    __syncthreads();

    // ==== Step 2+3: weights[c] = gate @ v[c];  out = f * (weights + bv*R) ====
    const unsigned* gu = (const unsigned*)gate_bf;
    const unsigned* vu = (const unsigned*)vB;
    const int c = wave;                         // one channel per wave
    v8f acc2[4] = {};
#pragma unroll
    for (int kb2 = 0; kb2 < 2; ++kb2) {
        union { v16bf v; unsigned u[8]; } Bf;   // B = v[c], N col = l16, K per element (table order)
#pragma unroll
        for (int e = 0; e < 16; ++e)
            Bf.v[e] = vB[c * VC + (kb2 * 32 + (e >> 3) * 16 + half * 8 + (e & 7)) * VK + l16];
#pragma unroll
        for (int ti = 0; ti < 4; ++ti) {
            union { v16bf v; unsigned u[8]; } A; // A = gate rows ti*16+l16
            int row = ti * 16 + l16;
#pragma unroll
            for (int r = 0; r < 8; ++r) {
                int K = (r >> 2) * 16 + half * 8 + (r & 3) * 2;
                A.u[r] = gu[(row * GSTR + kb2 * 32 + K) >> 1];
            }
            acc2[ti] = __builtin_amdgcn_wmma_f32_16x16x32_bf16(
                false, A.v, false, Bf.v, (short)0, acc2[ti], false, false);
        }
    }
    // epilogue: fp32 multiply with features, add bv*rowsum(gate)
    const float bvc = bv[s * C_ + c0 + c];
#pragma unroll
    for (int ti = 0; ti < 4; ++ti)
#pragma unroll
        for (int r = 0; r < 8; ++r) {
            int i = ti * 16 + r + half * 8;
            float wgt = acc2[ti][r] + bvc * Rrow[i];
            size_t off = fbase + (size_t)(c0 + c) * HW_ + i * W_ + j0 + l16;
            out[off] = f[off] * wgt;
        }
}

// ---------------------------------------------------------------------------
extern "C" void kernel_launch(void* const* d_in, const int* in_sizes, int n_in,
                              void* d_out, int out_size, void* d_ws, size_t ws_size,
                              hipStream_t stream) {
    const float* f  = (const float*)d_in[0];
    const float* Wq = (const float*)d_in[1];
    const float* bq = (const float*)d_in[2];
    const float* Wk = (const float*)d_in[3];
    const float* bk = (const float*)d_in[4];
    const float* Wv = (const float*)d_in[5];
    const float* bv = (const float*)d_in[6];
    float* out = (float*)d_out;
    float* ws  = (float*)d_ws;

    k_fbar<<<dim3(SB_ * C_), dim3(256), 0, stream>>>(f, ws);
    k_prep<<<dim3(SB_), dim3(256), 0, stream>>>(Wq, bq, Wk, bk, ws);
    k_gate<<<dim3(SB_ * 4), dim3(256), 0, stream>>>(f, ws);
    k_main<<<dim3(SB_, W_ / 16, C_ / 16), dim3(512), 0, stream>>>(f, Wv, bv, ws, out);
}
